// DSVTCrossAttentionBlock_48722109006386
// MI455X (gfx1250) — matrix operand order repaired and checked by reading, not compiled, not run on hardware
//
#include <hip/hip_runtime.h>

typedef __attribute__((ext_vector_type(16))) _Float16 v16h;
typedef __attribute__((ext_vector_type(8)))  float    v8f;

namespace {
constexpr int kNSets = 1280, kSetSz = 36, kNTok = 46080;
constexpr int kD = 192, kBD = 256, kM = 256, kH = 8, kDh = 24, kDhp = 32;
constexpr int kLp = 48, kDff = 2048;
constexpr int kQStride = kH * kDhp;              // 256
constexpr float kScale = 0.2041241452319315f;    // 24^-0.5
constexpr float kEps = 1e-5f;
}

#define WMMA(A, B, C) \
  __builtin_amdgcn_wmma_f32_16x16x32_f16(false, (A), false, (B), (short)0, (C), false, false)

#if __has_builtin(__builtin_amdgcn_global_load_async_to_lds_b128)
#define HAVE_ASYNC_LDS 1
typedef int v4i_b128 __attribute__((vector_size(16)));
typedef __attribute__((address_space(1))) v4i_b128* gptr_b128;  // global (printed __device__)
typedef __attribute__((address_space(3))) v4i_b128* lptr_b128;  // LDS
#endif

__device__ __forceinline__ void wait_asynccnt0() {
#if __has_builtin(__builtin_amdgcn_s_wait_asynccnt)
  __builtin_amdgcn_s_wait_asynccnt(0);
#else
  asm volatile("s_wait_asynccnt 0" ::: "memory");
#endif
}

// A fragment: 16x32 f16, row-major source with leading dim ld.
// Lane l (0-15): row l, K in {0..15}; lanes 16-31: row l-16, K in {16..31}.
__device__ __forceinline__ v16h ld_frag_a(const _Float16* p, int ld) {
  const int lane = threadIdx.x & 31;
  const int phf = lane >> 4, row = lane & 15;
  const _Float16* rp = p + (size_t)row * ld + phf * 16;
  v16h a;
#pragma unroll
  for (int j = 0; j < 8; ++j) a[j] = rp[j];
#pragma unroll
  for (int j = 0; j < 8; ++j) a[8 + j] = rp[8 + j];
  return a;
}

// B fragment for X @ W^T: B[k][n] = W[n][k]; p points at W row n0 + k-offset, row-major ld.
// Lane n holds column n; element j -> k = (lane>=16 ? 16 : 0) + j. Contiguous 32B per lane.
__device__ __forceinline__ v16h ld_frag_bt(const _Float16* p, int ld) {
  const int lane = threadIdx.x & 31;
  const _Float16* rp = p + (size_t)(lane & 15) * ld + (lane >> 4) * 16;
  v16h b;
#pragma unroll
  for (int j = 0; j < 16; ++j) b[j] = rp[j];
  return b;
}

// B fragment from a row-major K x N tile (direct, no transpose): B[k][n] = p[k*ld + n].
__device__ __forceinline__ v16h ld_frag_b(const _Float16* p, int ld) {
  const int lane = threadIdx.x & 31;
  const int phf = lane >> 4, col = lane & 15;
  v16h b;
#pragma unroll
  for (int j = 0; j < 16; ++j) b[j] = p[(size_t)(phf * 16 + j) * ld + col];
  return b;
}

// ---------------------------------------------------------------- prep kernels

__global__ void k_zero32(unsigned int* p, size_t n) {
  size_t i = (size_t)blockIdx.x * blockDim.x + threadIdx.x;
  if (i < n) p[i] = 0u;
}

__global__ void k_cvt16(const float* __restrict__ s, _Float16* __restrict__ d, int n) {
  int i = blockIdx.x * blockDim.x + threadIdx.x;
  if (i < n) d[i] = (_Float16)s[i];
}

__global__ void k_gather(const float* __restrict__ src, const float* __restrict__ pos,
                         const int* __restrict__ vinds, const int* __restrict__ vcrd,
                         _Float16* __restrict__ Xh, int* __restrict__ vb) {
  size_t idx = (size_t)blockIdx.x * blockDim.x + threadIdx.x;
  if (idx >= (size_t)kNTok * kD) return;
  int i = (int)(idx / kD), c = (int)(idx % kD);
  int vi = vinds[i];
  size_t p = (size_t)vi * kD + c;
  Xh[idx] = (_Float16)(src[p] + pos[p]);
  if (c == 0) vb[i] = vcrd[(size_t)vi * 4];
}

__global__ void k_box(const float* __restrict__ bf, const float* __restrict__ bp,
                      const int* __restrict__ bvc, _Float16* __restrict__ KinH,
                      _Float16* __restrict__ VinH, int* __restrict__ boxb) {
  int idx = blockIdx.x * blockDim.x + threadIdx.x;
  if (idx >= kM * kBD) return;
  KinH[idx] = (_Float16)(bf[idx] + bp[idx]);
  VinH[idx] = (_Float16)bf[idx];
  if (idx < kM) boxb[idx] = bvc[idx * 4];
}

// ------------------------------------------------------- K/V projection (tiny)
// K = (box+pos) @ Wk^T + bk ; V = box @ Wv^T + bv ; stored per-head, dh padded to 32.
__global__ void k_kv(const _Float16* __restrict__ KinH, const _Float16* __restrict__ VinH,
                     const _Float16* __restrict__ WkH, const _Float16* __restrict__ WvH,
                     const float* __restrict__ bk, const float* __restrict__ bv,
                     _Float16* __restrict__ Kh, _Float16* __restrict__ Vh) {
  int tile = blockIdx.x * 8 + (threadIdx.x >> 5);
  if (tile >= 16 * 12) return;
  int mt = tile / 12, ct = tile % 12;
  int m0 = mt * 16, c0 = ct * 16;
  int lane = threadIdx.x & 31, phf = lane >> 4, r = lane & 15;
  v8f ak = {}, av = {};
#pragma unroll
  for (int kt = 0; kt < 8; ++kt) {
    v16h aK = ld_frag_a(KinH + (size_t)m0 * kBD + kt * 32, kBD);
    v16h bK = ld_frag_bt(WkH + (size_t)c0 * kBD + kt * 32, kBD);
    ak = WMMA(aK, bK, ak);
    v16h aV = ld_frag_a(VinH + (size_t)m0 * kBD + kt * 32, kBD);
    v16h bV = ld_frag_bt(WvH + (size_t)c0 * kBD + kt * 32, kBD);
    av = WMMA(aV, bV, av);
  }
  int d = c0 + r, hh = d / kDh, dd = d % kDh;
  float bkv = bk[d], bvv = bv[d];
#pragma unroll
  for (int i = 0; i < 8; ++i) {
    int m = m0 + i + phf * 8;
    Kh[((size_t)hh * kM + m) * kDhp + dd] = (_Float16)(ak[i] + bkv);
    Vh[((size_t)hh * kM + m) * kDhp + dd] = (_Float16)(av[i] + bvv);
  }
}

// ------------------------------------------------------------------ Q projection
// Q = (src+pos)[inds] @ Wq^T + bq, scattered into per-set padded layout [set][48][h][32].
__global__ void k_qproj(const _Float16* __restrict__ Xh, const _Float16* __restrict__ WqH,
                        const float* __restrict__ bq, _Float16* __restrict__ Qp) {
  int tile = blockIdx.x * 8 + (threadIdx.x >> 5);
  if (tile >= (kNTok / 16) * 12) return;
  int rt = tile / 12, ct = tile % 12;
  int row0 = rt * 16, c0 = ct * 16;
  int lane = threadIdx.x & 31, phf = lane >> 4, r = lane & 15;
  v8f acc = {};
#pragma unroll
  for (int kt = 0; kt < 6; ++kt) {
    v16h a = ld_frag_a(Xh + (size_t)row0 * kD + kt * 32, kD);
    v16h b = ld_frag_bt(WqH + (size_t)c0 * kD + kt * 32, kD);
    acc = WMMA(a, b, acc);
  }
  int d = c0 + r, hh = d / kDh, dd = d % kDh;
  float bias = bq[d];
#pragma unroll
  for (int i = 0; i < 8; ++i) {
    int grow = row0 + i + phf * 8;
    int s = grow / kSetSz, l = grow % kSetSz;
    Qp[((size_t)s * kLp + l) * kQStride + hh * kDhp + dd] = (_Float16)(acc[i] + bias);
  }
}

// ------------------------------------------------------ attention (1 block/set, 1 wave/head)
__global__ __launch_bounds__(256) void k_attn(const _Float16* __restrict__ Qp,
                                              const _Float16* __restrict__ Kh,
                                              const _Float16* __restrict__ Vh,
                                              const int* __restrict__ vb,
                                              const int* __restrict__ boxb,
                                              _Float16* __restrict__ ctxh) {
  __shared__ _Float16 qtile[kLp * kQStride];  // 24 KB: this set's padded Q (read 3x per wave)
  __shared__ _Float16 probs[kH][16][128];     // 32 KB: per-wave softmax staging
  int s = blockIdx.x;
  int h = threadIdx.x >> 5;
  int lane = threadIdx.x & 31, phf = lane >> 4, r = lane & 15;
  _Float16* pl = &probs[h][0][0];

  // Stage Q[set] into LDS: 48*256 halves = 1536 x b128, 6 chunks per thread.
  const _Float16* qsrc = Qp + (size_t)s * kLp * kQStride;
#ifdef HAVE_ASYNC_LDS
#pragma unroll
  for (int t = 0; t < 6; ++t) {
    int e = (t * 256 + (int)threadIdx.x) * 8;  // element offset, 16B granules
    __builtin_amdgcn_global_load_async_to_lds_b128(
        (gptr_b128)(qsrc + e), (lptr_b128)(qtile + e), 0, 0);
  }
  wait_asynccnt0();
#else
  for (int t = threadIdx.x; t < kLp * kQStride / 8; t += 256)
    ((float4*)qtile)[t] = ((const float4*)qsrc)[t];
#endif
  __syncthreads();

  int boxbv[16];
#pragma unroll
  for (int mt = 0; mt < 16; ++mt) boxbv[mt] = boxb[mt * 16 + r];

  for (int rt = 0; rt < 3; ++rt) {
    int vbrow[8];
#pragma unroll
    for (int i = 0; i < 8; ++i) {
      int l = rt * 16 + i + phf * 8;
      vbrow[i] = (l < kSetSz) ? vb[s * kSetSz + l] : 0x7fffffff;  // pad rows: fully masked
    }
    v16h aq = ld_frag_a(qtile + (rt * 16) * kQStride + h * kDhp, kQStride);
    v8f acc[16] = {};
#pragma unroll
    for (int mt = 0; mt < 16; ++mt) {
      v16h bk_ = ld_frag_bt(Kh + ((size_t)h * kM + mt * 16) * kDhp, kDhp);
      acc[mt] = WMMA(aq, bk_, acc[mt]);
    }
    // masked softmax per row; reductions across the 16 lanes of each half (wave32)
#pragma unroll
    for (int i = 0; i < 8; ++i) {
      float mx = -INFINITY;
#pragma unroll
      for (int mt = 0; mt < 16; ++mt)
        if (vbrow[i] == boxbv[mt]) mx = fmaxf(mx, acc[mt][i] * kScale);
#pragma unroll
      for (int m2 = 1; m2 <= 8; m2 <<= 1) mx = fmaxf(mx, __shfl_xor(mx, m2, 32));
      float sum = 0.f;
      float ev[16];
#pragma unroll
      for (int mt = 0; mt < 16; ++mt) {
        float e = (vbrow[i] == boxbv[mt] && mx > -INFINITY)
                      ? __expf(acc[mt][i] * kScale - mx) : 0.f;
        ev[mt] = e;
        sum += e;
      }
#pragma unroll
      for (int m2 = 1; m2 <= 8; m2 <<= 1) sum += __shfl_xor(sum, m2, 32);
      float inv = sum > 0.f ? 1.f / sum : 0.f;  // fully-masked row -> 0 (nan_to_num)
#pragma unroll
      for (int mt = 0; mt < 16; ++mt) acc[mt][i] = ev[mt] * inv;
    }
    // ctx = P @ V_h, probs restaged via LDS in two 128-key half-passes
    v8f cacc[2] = {};
#pragma unroll
    for (int hp = 0; hp < 2; ++hp) {
#pragma unroll
      for (int i = 0; i < 8; ++i) {
        int row = i + phf * 8;
#pragma unroll
        for (int m2 = 0; m2 < 8; ++m2)
          pl[row * 128 + m2 * 16 + r] = (_Float16)acc[hp * 8 + m2][i];
      }
#pragma unroll
      for (int nt = 0; nt < 2; ++nt) {
#pragma unroll
        for (int k2 = 0; k2 < 4; ++k2) {
          v16h pa = ld_frag_a(pl + k2 * 32, 128);
          v16h pb = ld_frag_b(Vh + ((size_t)h * kM + hp * 128 + k2 * 32) * kDhp + nt * 16, kDhp);
          cacc[nt] = WMMA(pa, pb, cacc[nt]);
        }
      }
    }
#pragma unroll
    for (int nt = 0; nt < 2; ++nt) {
#pragma unroll
      for (int i = 0; i < 8; ++i) {
        int l = rt * 16 + i + phf * 8;
        int d = nt * 16 + r;
        if (l < kSetSz && d < kDh)
          ctxh[((size_t)s * kSetSz + l) * kD + h * kDh + d] = (_Float16)cacc[nt][i];
      }
    }
  }
}

// --------------------------------------- out-proj + residual + LayerNorm1 (fused)
__global__ void k_oproj(const _Float16* __restrict__ ctxh, const _Float16* __restrict__ WoH,
                        const float* __restrict__ bo, const float* __restrict__ src,
                        const int* __restrict__ vinds, const float* __restrict__ g1,
                        const float* __restrict__ b1ln, float* __restrict__ xf,
                        _Float16* __restrict__ xh) {
  int tile = blockIdx.x * 8 + (threadIdx.x >> 5);
  if (tile >= kNTok / 16) return;
  int row0 = tile * 16;
  int lane = threadIdx.x & 31, phf = lane >> 4, r = lane & 15;
  v8f acc[12] = {};
#pragma unroll
  for (int kt = 0; kt < 6; ++kt) {
    v16h a = ld_frag_a(ctxh + (size_t)row0 * kD + kt * 32, kD);
#pragma unroll
    for (int ct = 0; ct < 12; ++ct) {
      v16h b = ld_frag_bt(WoH + (size_t)(ct * 16) * kD + kt * 32, kD);
      acc[ct] = WMMA(a, b, acc[ct]);
    }
  }
  float bov[12], g1v[12], b1v[12];
#pragma unroll
  for (int ct = 0; ct < 12; ++ct) {
    bov[ct] = bo[ct * 16 + r];
    g1v[ct] = g1[ct * 16 + r];
    b1v[ct] = b1ln[ct * 16 + r];
  }
#pragma unroll
  for (int i = 0; i < 8; ++i) {
    int grow = row0 + i + phf * 8;
    int vi = vinds[grow];
    float s1 = 0.f, s2 = 0.f;
#pragma unroll
    for (int ct = 0; ct < 12; ++ct) {
      float v = acc[ct][i] + bov[ct];
      if (v != v) v = 0.f;                       // nan_to_num
      else if (v > 3.0e38f) v = 3.4028234e38f;
      else if (v < -3.0e38f) v = -3.4028234e38f;
      v += src[(size_t)vi * kD + ct * 16 + r];   // residual
      acc[ct][i] = v;
      s1 += v;
      s2 += v * v;
    }
#pragma unroll
    for (int m2 = 1; m2 <= 8; m2 <<= 1) {
      s1 += __shfl_xor(s1, m2, 32);
      s2 += __shfl_xor(s2, m2, 32);
    }
    float mean = s1 * (1.f / kD);
    float var = s2 * (1.f / kD) - mean * mean;
    float rstd = rsqrtf(var + kEps);
#pragma unroll
    for (int ct = 0; ct < 12; ++ct) {
      float xv = (acc[ct][i] - mean) * rstd * g1v[ct] + b1v[ct];
      xf[(size_t)grow * kD + ct * 16 + r] = xv;
      xh[(size_t)grow * kD + ct * 16 + r] = (_Float16)xv;
    }
  }
}

// ---------------- FFN (both GEMMs fused, hidden kept in LDS) + residual + LN2 + scatter-out
__global__ __launch_bounds__(256) void k_ffn(const _Float16* __restrict__ xh,
                                             const float* __restrict__ xf,
                                             const _Float16* __restrict__ W1H,
                                             const _Float16* __restrict__ W2H,
                                             const float* __restrict__ b1g,
                                             const float* __restrict__ b2g,
                                             const int* __restrict__ vinds,
                                             const float* __restrict__ g2,
                                             const float* __restrict__ b2ln,
                                             float* __restrict__ out) {
  __shared__ _Float16 hid[16 * 1024];  // 32 KB, 1024-column half of the hidden tile
  __shared__ float ybuf[16 * kD];      // 12 KB
  __shared__ float mu[16], rs[16];
  int row0 = blockIdx.x * 16;
  int wid = threadIdx.x >> 5;
  int lane = threadIdx.x & 31, phf = lane >> 4, r = lane & 15;
  v16h ax[6];
#pragma unroll
  for (int kt = 0; kt < 6; ++kt) ax[kt] = ld_frag_a(xh + (size_t)row0 * kD + kt * 32, kD);
  v8f yacc[2] = {};
  for (int pass = 0; pass < 2; ++pass) {
    // GEMM1: hidden columns [pass*1024, +1024), 8 col-tiles per wave
#pragma unroll
    for (int t = 0; t < 8; ++t) {
      int ln0 = (wid * 8 + t) * 16;
      int n0 = pass * 1024 + ln0;
      v8f acc = {};
#pragma unroll
      for (int kt = 0; kt < 6; ++kt) {
        v16h b = ld_frag_bt(W1H + (size_t)n0 * kD + kt * 32, kD);
        acc = WMMA(ax[kt], b, acc);
      }
      if (t < 7)  // L2-latency cover for the next W1 column tile
        __builtin_prefetch(W1H + (size_t)(n0 + 16) * kD, 0, 1);
      float bb = b1g[n0 + r];
#pragma unroll
      for (int i = 0; i < 8; ++i) {
        float v = acc[i] + bb;
        hid[(i + phf * 8) * 1024 + ln0 + r] = (_Float16)(v > 0.f ? v : 0.f);  // ReLU
      }
    }
    __syncthreads();
    // GEMM2 partial accumulation; shared A-fragment feeds both output tiles of this wave
    int c0 = wid * 16, c1 = (wid + 8) * 16;
    for (int kt = 0; kt < 32; ++kt) {
      v16h pa = ld_frag_a(hid + kt * 32, 1024);
      v16h pb0 = ld_frag_bt(W2H + (size_t)c0 * kDff + pass * 1024 + kt * 32, kDff);
      yacc[0] = WMMA(pa, pb0, yacc[0]);
      if (kt < 31)
        __builtin_prefetch(W2H + (size_t)c0 * kDff + pass * 1024 + (kt + 1) * 32, 0, 1);
      if (wid < 4) {
        v16h pb1 = ld_frag_bt(W2H + (size_t)c1 * kDff + pass * 1024 + kt * 32, kDff);
        yacc[1] = WMMA(pa, pb1, yacc[1]);
      }
    }
    __syncthreads();
  }
#pragma unroll
  for (int u = 0; u < 2; ++u) {
    int ct = wid + u * 8;
    if (ct < 12) {
      float bb = b2g[ct * 16 + r];
#pragma unroll
      for (int i = 0; i < 8; ++i) {
        int row = i + phf * 8;
        ybuf[row * kD + ct * 16 + r] =
            yacc[u][i] + bb + xf[(size_t)(row0 + row) * kD + ct * 16 + r];
      }
    }
  }
  __syncthreads();
  if (threadIdx.x < 16) {
    float s1 = 0.f, s2 = 0.f;
    for (int c = 0; c < kD; ++c) {
      float v = ybuf[threadIdx.x * kD + c];
      s1 += v;
      s2 += v * v;
    }
    float mean = s1 * (1.f / kD);
    float var = s2 * (1.f / kD) - mean * mean;
    mu[threadIdx.x] = mean;
    rs[threadIdx.x] = rsqrtf(var + kEps);
  }
  __syncthreads();
  for (int idx = threadIdx.x; idx < 16 * kD; idx += 256) {
    int row = idx / kD, c = idx % kD;
    int vi = vinds[row0 + row];
    out[(size_t)vi * kD + c] = (ybuf[idx] - mu[row]) * rs[row] * g2[c] + b2ln[c];
  }
}

// -------------------------------------------------------------------------- host

extern "C" void kernel_launch(void* const* d_in, const int* in_sizes, int n_in,
                              void* d_out, int out_size, void* d_ws, size_t ws_size,
                              hipStream_t stream) {
  (void)in_sizes; (void)n_in; (void)out_size; (void)ws_size;
  const float* src = (const float*)d_in[0];
  const int* vcrd = (const int*)d_in[1];
  const float* boxf = (const float*)d_in[2];
  const int* bvc = (const int*)d_in[3];
  const float* pos = (const float*)d_in[4];
  const float* boxp = (const float*)d_in[5];
  const int* vinds = (const int*)d_in[6];
  const float* Wq = (const float*)d_in[7];
  const float* bq = (const float*)d_in[8];
  const float* Wk = (const float*)d_in[9];
  const float* bk = (const float*)d_in[10];
  const float* Wv = (const float*)d_in[11];
  const float* bv = (const float*)d_in[12];
  const float* Wo = (const float*)d_in[13];
  const float* bo = (const float*)d_in[14];
  const float* W1 = (const float*)d_in[15];
  const float* b1 = (const float*)d_in[16];
  const float* W2 = (const float*)d_in[17];
  const float* b2 = (const float*)d_in[18];
  const float* g1 = (const float*)d_in[19];
  const float* bln1 = (const float*)d_in[20];
  const float* g2 = (const float*)d_in[21];
  const float* bln2 = (const float*)d_in[22];
  float* out = (float*)d_out;

  char* ws = (char*)d_ws;
  size_t off = 0;
  auto take = [&](size_t bytes) -> void* {
    off = (off + 255) & ~(size_t)255;
    void* p = ws + off;
    off += bytes;
    return p;
  };
  _Float16* Xh   = (_Float16*)take((size_t)kNTok * kD * 2);
  _Float16* WqH  = (_Float16*)take((size_t)kD * kD * 2);
  _Float16* WkH  = (_Float16*)take((size_t)kD * kBD * 2);
  _Float16* WvH  = (_Float16*)take((size_t)kD * kBD * 2);
  _Float16* WoH  = (_Float16*)take((size_t)kD * kD * 2);
  _Float16* W1H  = (_Float16*)take((size_t)kDff * kD * 2);
  _Float16* W2H  = (_Float16*)take((size_t)kD * kDff * 2);
  _Float16* KinH = (_Float16*)take((size_t)kM * kBD * 2);
  _Float16* VinH = (_Float16*)take((size_t)kM * kBD * 2);
  _Float16* Kh   = (_Float16*)take((size_t)kH * kM * kDhp * 2);
  _Float16* Vh   = (_Float16*)take((size_t)kH * kM * kDhp * 2);
  _Float16* Qp   = (_Float16*)take((size_t)kNSets * kLp * kQStride * 2);
  _Float16* ctxh = (_Float16*)take((size_t)kNTok * kD * 2);
  float*    xf   = (float*)take((size_t)kNTok * kD * 4);
  _Float16* xhb  = (_Float16*)take((size_t)kNTok * kD * 2);
  int*      vb   = (int*)take((size_t)kNTok * 4);
  int*      boxb = (int*)take((size_t)kM * 4);

  auto cdiv = [](size_t a, size_t b) { return (unsigned)((a + b - 1) / b); };

  // zero the padded staging buffers (deterministic every call)
  size_t nQpW = (size_t)kNSets * kLp * kQStride / 2;
  k_zero32<<<cdiv(nQpW, 256), 256, 0, stream>>>((unsigned int*)Qp, nQpW);
  size_t nKhW = (size_t)kH * kM * kDhp / 2;
  k_zero32<<<cdiv(nKhW, 256), 256, 0, stream>>>((unsigned int*)Kh, nKhW);
  k_zero32<<<cdiv(nKhW, 256), 256, 0, stream>>>((unsigned int*)Vh, nKhW);

  // f16 staging
  k_gather<<<cdiv((size_t)kNTok * kD, 256), 256, 0, stream>>>(src, pos, vinds, vcrd, Xh, vb);
  k_box<<<cdiv((size_t)kM * kBD, 256), 256, 0, stream>>>(boxf, boxp, bvc, KinH, VinH, boxb);
  k_cvt16<<<cdiv(kD * kD, 256), 256, 0, stream>>>(Wq, WqH, kD * kD);
  k_cvt16<<<cdiv(kD * kBD, 256), 256, 0, stream>>>(Wk, WkH, kD * kBD);
  k_cvt16<<<cdiv(kD * kBD, 256), 256, 0, stream>>>(Wv, WvH, kD * kBD);
  k_cvt16<<<cdiv(kD * kD, 256), 256, 0, stream>>>(Wo, WoH, kD * kD);
  k_cvt16<<<cdiv(kDff * kD, 256), 256, 0, stream>>>(W1, W1H, kDff * kD);
  k_cvt16<<<cdiv(kD * kDff, 256), 256, 0, stream>>>(W2, W2H, kD * kDff);

  // projections
  k_kv<<<24, 256, 0, stream>>>(KinH, VinH, WkH, WvH, bk, bv, Kh, Vh);
  k_qproj<<<(kNTok / 16) * 12 / 8, 256, 0, stream>>>(Xh, WqH, bq, Qp);

  // attention (1 block per set, 1 wave per head)
  k_attn<<<kNSets, 256, 0, stream>>>(Qp, Kh, Vh, vb, boxb, ctxh);

  // out-proj + residual + LN1
  k_oproj<<<(kNTok / 16) / 8, 256, 0, stream>>>(ctxh, WoH, bo, src, vinds, g1, bln1, xf, xhb);

  // fused FFN + residual + LN2 + scatter to output
  k_ffn<<<kNTok / 16, 256, 0, stream>>>(xhb, xf, W1H, W2H, b1, b2, vinds, g2, bln2, out);
}
